// DrugResponsePrior_82360292868295
// MI455X (gfx1250) — compile-verified, matrix operand order checked by reading, not run on hardware
//
#include <hip/hip_runtime.h>
#include <hip/hip_bf16.h>
#include <math.h>

typedef __attribute__((ext_vector_type(16))) _Float16 v16h;
typedef __attribute__((ext_vector_type(8)))  float    v8f;
typedef __attribute__((ext_vector_type(4)))  unsigned int u32x4;
typedef int i32x4 __attribute__((vector_size(16)));

#define N_CELLS   900
#define N_MISSING 100
#define N_DRUGS   256
#define NFEAT     1024
#define CELL_EMB  1024
#define DRUG_EMB  128
#define HID       200
#define NDOSES    9
#define HPAD      224   // K padded to 7*32
#define NPAD      208   // N padded to 13*16

#define W2S_STRIDE 232               // halfs per row in staged Wf2 (bank-friendly)
#define W2S_HALFS  (NPAD * W2S_STRIDE)   // 48256
#define W2S_BYTES  (W2S_HALFS * 2)       // 96512
#define W2S_CHUNKS (W2S_BYTES / 16)      // 6032 x b128

// ---------------- workspace layout (bytes) ----------------
#define OFF_CELLN 0            // 200*1024 f32 = 819200
#define OFF_DRUGN 819200       // 256*128  f32 = 131072
#define OFF_C1    950272       // 200*200  f32 = 160000
#define OFF_D1    1110272      // 256*200  f32 = 204800
#define OFF_W2T   1315072      // 208*232  f16 = 96512
#define OFF_W3T   1411584      // 16*224   f16 = 7168
// total 1418752 bytes

// ---- K1: cellN[0:100] = l2norm(relu(cell_features[s]@W1+b1)),
//          cellN[100:200] = l2norm(missing_emb[s]) ------------------------
__global__ __launch_bounds__(256) void k_cellN(const float* __restrict__ cf,
                                               const float* __restrict__ memb,
                                               const float* __restrict__ W1,
                                               const float* __restrict__ b1,
                                               float* __restrict__ cellN) {
  __shared__ float red[256];
  int bid = blockIdx.x, t = threadIdx.x;
  float v0, v1, v2, v3;
  int c0 = t, c1 = t + 256, c2 = t + 512, c3 = t + 768;
  if (bid < N_MISSING) {
    const float* frow = cf + (size_t)bid * NFEAT;
    float a0 = b1[c0], a1 = b1[c1], a2 = b1[c2], a3 = b1[c3];
    for (int k = 0; k < NFEAT; ++k) {
      float f = frow[k];
      const float* w = W1 + (size_t)k * CELL_EMB;
      a0 += f * w[c0]; a1 += f * w[c1]; a2 += f * w[c2]; a3 += f * w[c3];
    }
    v0 = fmaxf(a0, 0.f); v1 = fmaxf(a1, 0.f); v2 = fmaxf(a2, 0.f); v3 = fmaxf(a3, 0.f);
  } else {
    const float* mrow = memb + (size_t)(bid - N_MISSING) * CELL_EMB;
    v0 = mrow[c0]; v1 = mrow[c1]; v2 = mrow[c2]; v3 = mrow[c3];
  }
  red[t] = v0 * v0 + v1 * v1 + v2 * v2 + v3 * v3;
  __syncthreads();
  for (int s = 128; s > 0; s >>= 1) {
    if (t < s) red[t] += red[t + s];
    __syncthreads();
  }
  float scale = 1.f / fmaxf(sqrtf(red[0]), 1e-12f);
  float* outr = cellN + (size_t)bid * CELL_EMB;
  outr[c0] = v0 * scale; outr[c1] = v1 * scale; outr[c2] = v2 * scale; outr[c3] = v3 * scale;
}

// ---- K2: drugN = l2norm(drug_emb) rows -----------------------------------
__global__ __launch_bounds__(128) void k_drugN(const float* __restrict__ demb,
                                               float* __restrict__ drugN) {
  __shared__ float red[128];
  int bid = blockIdx.x, t = threadIdx.x;
  float v = demb[(size_t)bid * DRUG_EMB + t];
  red[t] = v * v;
  __syncthreads();
  for (int s = 64; s > 0; s >>= 1) {
    if (t < s) red[t] += red[t + s];
    __syncthreads();
  }
  float scale = 1.f / fmaxf(sqrtf(red[0]), 1e-12f);
  drugN[(size_t)bid * DRUG_EMB + t] = v * scale;
}

// ---- K3: C1[i][c] = cellN[i] . Wf1[0:1024, c] + bf1[c] -------------------
__global__ __launch_bounds__(256) void k_C1(const float* __restrict__ cellN,
                                            const float* __restrict__ Wf1,
                                            const float* __restrict__ bf1,
                                            float* __restrict__ C1) {
  int i = blockIdx.x, c = threadIdx.x;
  if (c >= HID) return;
  const float* row = cellN + (size_t)i * CELL_EMB;
  float acc = bf1[c];
  for (int k = 0; k < CELL_EMB; ++k) acc += row[k] * Wf1[(size_t)k * HID + c];
  C1[(size_t)i * HID + c] = acc;
}

// ---- K4: D1[d][c] = drugN[d] . Wf1[1024:1152, c] -------------------------
__global__ __launch_bounds__(256) void k_D1(const float* __restrict__ drugN,
                                            const float* __restrict__ Wf1,
                                            float* __restrict__ D1) {
  int d = blockIdx.x, c = threadIdx.x;
  if (c >= HID) return;
  const float* row = drugN + (size_t)d * DRUG_EMB;
  float acc = 0.f;
  for (int k = 0; k < DRUG_EMB; ++k) acc += row[k] * Wf1[(size_t)(CELL_EMB + k) * HID + c];
  D1[(size_t)d * HID + c] = acc;
}

// ---- K5: pack Wf2 -> f16 [n=208][k=232] (zero padded), Wf3 -> [16][224] --
__global__ __launch_bounds__(232) void k_pack(const float* __restrict__ Wf2,
                                              const float* __restrict__ Wf3,
                                              _Float16* __restrict__ W2T,
                                              _Float16* __restrict__ W3T) {
  int bid = blockIdx.x, k = threadIdx.x;
  if (bid < NPAD) {
    int n = bid;
    float v = (n < HID && k < HID) ? Wf2[(size_t)k * HID + n] : 0.f;
    W2T[(size_t)n * W2S_STRIDE + k] = (_Float16)v;
  } else if (k < HPAD) {
    int j = bid - NPAD;
    float v = (j < NDOSES && k < HID) ? Wf3[(size_t)k * NDOSES + j] : 0.f;
    W3T[(size_t)j * HPAD + k] = (_Float16)v;
  }
}

// ---- main: stage Wf2 in LDS once per block (async copy), then per-wave
//      16-sample tiles: h1 -> WMMA layer2 -> WMMA layer3 -> softplus/cumsum
__global__ __launch_bounds__(256) void k_main(const int* __restrict__ idx,
                                              const int* __restrict__ tidx,
                                              const int* __restrict__ ismiss,
                                              const int* __restrict__ cmap,
                                              const int* __restrict__ dmap,
                                              const float* __restrict__ C1,
                                              const float* __restrict__ D1,
                                              const char* __restrict__ W2Tg,    // staged f16 source
                                              const unsigned int* __restrict__ W3T, // f16 pairs
                                              const float* __restrict__ bf2,
                                              const float* __restrict__ bf3,
                                              float* __restrict__ out, int B) {
  __shared__ __align__(16) _Float16 W2s[W2S_HALFS];     // 96512 B, block-shared B matrix
  __shared__ __align__(16) _Float16 hbuf[8][16 * HPAD]; // per-wave 16x224 f16 tile
  __shared__ int ids[8][32];                            // per-wave 16 cell + 16 drug ids

  const int lane = threadIdx.x & 31;
  const int wave = threadIdx.x >> 5;
  const int base = blockIdx.x * 128 + wave * 16;
  const int m    = lane & 15;
  const int hi8  = (lane >= 16) ? 8 : 0;

  // gather per-sample (cell_id, drug_id)
  if (lane < 16) {
    int s = base + lane, cid = 0, did = 0;
    if (s < B) {
      int i  = idx[s];
      int ti = tidx[s];
      int cm = cmap[i];
      cid = (ismiss[i] != 0) ? (N_MISSING + cm) : cm;
      did = dmap[ti];
    }
    ids[wave][lane]      = cid;
    ids[wave][16 + lane] = did;
  }

  // ---- stage Wf2 (f16, padded) into LDS: async DMA path if available ----
#if defined(__AMDGCN__) && __has_builtin(__builtin_amdgcn_global_load_async_to_lds_b128) && __has_builtin(__builtin_amdgcn_s_wait_asynccnt)
  typedef __attribute__((address_space(1))) i32x4 GI4;
  typedef __attribute__((address_space(3))) i32x4 LI4;
  for (int i = threadIdx.x; i < W2S_CHUNKS; i += 256) {
    __builtin_amdgcn_global_load_async_to_lds_b128(
        (GI4*)(W2Tg + (size_t)i * 16),
        (LI4*)((char*)W2s + (size_t)i * 16),
        0, 0);
  }
  __builtin_amdgcn_s_wait_asynccnt(0);
#else
  {
    const u32x4* s4 = (const u32x4*)W2Tg;
    u32x4*       d4 = (u32x4*)W2s;
    for (int i = threadIdx.x; i < W2S_CHUNKS; i += 256) d4[i] = s4[i];
  }
#endif
  __syncthreads();

  _Float16* h1 = hbuf[wave];
  // h1[r][c] = relu(C1[cid][c] + D1[did][c]), zero-padded to 224 cols
  for (int r = 0; r < 16; ++r) {
    int cid = ids[wave][r], did = ids[wave][16 + r];
    const float* c1r = C1 + (size_t)cid * HID;
    const float* d1r = D1 + (size_t)did * HID;
#pragma unroll
    for (int i = 0; i < 7; ++i) {
      int c = lane + i * 32;
      float v = (c < HID) ? fmaxf(c1r[c] + d1r[c], 0.f) : 0.f;
      h1[r * HPAD + c] = (_Float16)v;
    }
  }
  __syncthreads();

  // A fragments: 16-bit A 16x32 layout (lane -> M = lane&15; K pairs per VGPR)
  typedef union { v16h h; unsigned int u[8]; } frag;
  frag A[7];
  const unsigned int* h1u  = (const unsigned int*)h1;
  const unsigned int* W2su = (const unsigned int*)W2s;
#pragma unroll
  for (int kt = 0; kt < 7; ++kt)
#pragma unroll
    for (int v = 0; v < 8; ++v) {
      int k = kt * 32 + ((v & 4) ? 16 : 0) + hi8 + (v & 3) * 2;
      A[kt].u[v] = h1u[(m * HPAD + k) >> 1];
    }
  __syncthreads();

  // ---- layer 2: h2 = relu(h1 @ Wf2 + bf2), 13 N-tiles x 7 K-tiles WMMA ----
  for (int nt = 0; nt < 13; ++nt) {
    int n = nt * 16 + m;
    float binit = (n < HID) ? bf2[n] : 0.f;
    v8f acc;
#pragma unroll
    for (int v = 0; v < 8; ++v) acc[v] = binit;
#pragma unroll
    for (int kt = 0; kt < 7; ++kt) {
      frag Bf;
#pragma unroll
      for (int v = 0; v < 8; ++v) {
        int k = kt * 32 + ((v & 4) ? 16 : 0) + hi8 + (v & 3) * 2;
        Bf.u[v] = W2su[(n * W2S_STRIDE + k) >> 1];
      }
      acc = __builtin_amdgcn_wmma_f32_16x16x32_f16(false, A[kt].h, false, Bf.h,
                                                   (short)0, acc, false, false);
    }
    // relu -> f16 back into the (reused) LDS tile as layer-3 A operand
#pragma unroll
    for (int v = 0; v < 8; ++v)
      h1[(v + hi8) * HPAD + n] = (_Float16)fmaxf(acc[v], 0.f);
  }
  if (lane < 16) { // zero pad cols 208..223
    for (int r = 0; r < 16; ++r) h1[r * HPAD + NPAD + lane] = (_Float16)0.f;
  }
  __syncthreads();

  // reload A fragments (now h2)
#pragma unroll
  for (int kt = 0; kt < 7; ++kt)
#pragma unroll
    for (int v = 0; v < 8; ++v) {
      int k = kt * 32 + ((v & 4) ? 16 : 0) + hi8 + (v & 3) * 2;
      A[kt].u[v] = h1u[(m * HPAD + k) >> 1];
    }
  __syncthreads();

  // ---- layer 3: f = h2 @ Wf3 + bf3 (single 16-wide N tile, 9 valid) ----
  v8f acc3;
  float b3 = (m < NDOSES) ? bf3[m] : 0.f;
#pragma unroll
  for (int v = 0; v < 8; ++v) acc3[v] = b3;
#pragma unroll
  for (int kt = 0; kt < 7; ++kt) {
    frag Bf;
#pragma unroll
    for (int v = 0; v < 8; ++v) {
      int k = kt * 32 + ((v & 4) ? 16 : 0) + hi8 + (v & 3) * 2;
      Bf.u[v] = W3T[((size_t)m * HPAD + k) >> 1];
    }
    acc3 = __builtin_amdgcn_wmma_f32_16x16x32_f16(false, A[kt].h, false, Bf.h,
                                                  (short)0, acc3, false, false);
  }

  // scatter D tile to LDS (reuse buffer as f32), then per-row epilogue
  float* ftile = (float*)h1;
#pragma unroll
  for (int v = 0; v < 8; ++v) ftile[(v + hi8) * 16 + m] = acc3[v];
  __syncthreads();

  if (lane < 16) {
    int s = base + lane;
    if (s < B) {
      float mu = ftile[lane * 16 + 0];
      out[(size_t)s * NDOSES + 0] = mu;
#pragma unroll
      for (int j = 1; j < NDOSES; ++j) {
        float x  = ftile[lane * 16 + j];
        float sp = (x > 0.f) ? (x + log1pf(expf(-x))) : log1pf(expf(x));
        mu += sp;
        out[(size_t)s * NDOSES + j] = mu;
      }
    }
  }
}

extern "C" void kernel_launch(void* const* d_in, const int* in_sizes, int n_in,
                              void* d_out, int out_size, void* d_ws, size_t ws_size,
                              hipStream_t stream) {
  const int*   idx    = (const int*)d_in[0];
  const int*   tidx   = (const int*)d_in[1];
  const int*   ismiss = (const int*)d_in[2];
  const int*   cmap   = (const int*)d_in[3];
  const int*   dmap   = (const int*)d_in[4];
  const float* cf     = (const float*)d_in[5];
  const float* memb   = (const float*)d_in[6];
  const float* demb   = (const float*)d_in[7];
  const float* W1     = (const float*)d_in[8];
  const float* b1     = (const float*)d_in[9];
  const float* Wf1    = (const float*)d_in[10];
  const float* bf1    = (const float*)d_in[11];
  const float* Wf2    = (const float*)d_in[12];
  const float* bf2    = (const float*)d_in[13];
  const float* Wf3    = (const float*)d_in[14];
  const float* bf3    = (const float*)d_in[15];
  float* out = (float*)d_out;
  int B = in_sizes[0];

  char* ws = (char*)d_ws;
  float*     cellN = (float*)(ws + OFF_CELLN);
  float*     drugN = (float*)(ws + OFF_DRUGN);
  float*     C1    = (float*)(ws + OFF_C1);
  float*     D1    = (float*)(ws + OFF_D1);
  _Float16*  W2T   = (_Float16*)(ws + OFF_W2T);
  _Float16*  W3T   = (_Float16*)(ws + OFF_W3T);

  k_cellN<<<200, 256, 0, stream>>>(cf, memb, W1, b1, cellN);
  k_drugN<<<N_DRUGS, 128, 0, stream>>>(demb, drugN);
  k_C1<<<200, 256, 0, stream>>>(cellN, Wf1, bf1, C1);
  k_D1<<<N_DRUGS, 256, 0, stream>>>(drugN, Wf1, D1);
  k_pack<<<NPAD + 16, W2S_STRIDE, 0, stream>>>(Wf2, Wf3, W2T, W3T);

  int blocks = (B + 127) / 128;
  k_main<<<blocks, 256, 0, stream>>>(idx, tidx, ismiss, cmap, dmap, C1, D1,
                                     (const char*)W2T, (const unsigned int*)W3T,
                                     bf2, bf3, out, B);
}
